// HRMChess_28355374088988
// MI455X (gfx1250) — compile-verified, matrix operand order checked by reading, not run on hardware
//
#include <hip/hip_runtime.h>
#include <hip/hip_bf16.h>

// ---------------------------------------------------------------------------
// HRM-Chess forward for MI455X (gfx1250): wave32, all GEMMs on
// v_wmma_f32_16x16x32_bf16 (bf16 operands, f32 accumulation).
//
// ~200 GFLOP of dense GEMM vs ~250 MB of traffic -> compute bound; working
// set after the conv front-end is L2-resident (192 MB), so the matrix pipe is
// the limiter. Each wave owns a 32x64 C tile (8 WMMAs per K-step of 32).
// For K <= 512 (all encoder GEMMs) the 64-row weight strip (32-64 KB) is
// staged into LDS by one Tensor Data Mover descriptor (tensor_load_to_lds +
// s_wait_tensorcnt), so the K-loop's B fragments come from DS instead of VMEM.
// Large-K GEMMs (conv2 K=2304, enh K=16384) stream from global with
// global_prefetch_b8 on the A stream.
// ---------------------------------------------------------------------------

typedef __hip_bfloat16 bf16;
typedef __attribute__((ext_vector_type(16))) __bf16 v16bf;
typedef __attribute__((ext_vector_type(8)))  float  v8f;
typedef __attribute__((ext_vector_type(4)))  unsigned int v4u;
typedef __attribute__((ext_vector_type(8)))  int v8i;
typedef __attribute__((ext_vector_type(4)))  int v4i;

static constexpr int NHEADC = 4;
static constexpr int SEQC   = 64;
static constexpr int BATCHC = 512;
static constexpr int DFFC   = 512;

// ----------------------------- device helpers ------------------------------

__device__ inline float bf2f(bf16 v) { return __bfloat162float(v); }
__device__ inline bf16  f2bf(float v) { return __float2bfloat16(v); }

__device__ inline float gelu_f(float x) {
  return 0.5f * x * (1.0f + erff(x * 0.70710678118654752f));
}

// 16-bit A/B fragment (ISA 7.12.2): lane m = lane&15, half = lane>>4.
// Elements 0..7  hold K = k0 + half*8 + (0..7)        (contiguous 16 bytes)
// Elements 8..15 hold K = k0 + 16 + half*8 + (0..7)   (contiguous 16 bytes)
// B (weights, row-major [N,K]) mirrors this with n = lane&15.
__device__ inline v16bf load_frag16(const bf16* __restrict__ base, int ld,
                                    int row, int k0, int half) {
  const bf16* p = base + (size_t)row * ld + k0 + half * 8;
  v16bf f;
  ((float4*)&f)[0] = *(const float4*)(p);
  ((float4*)&f)[1] = *(const float4*)(p + 16);
  return f;
}

// ----------------------------- WMMA GEMM -----------------------------------
// C[M,N] = act(A[M,K] * W[N,K]^T + bias[N]).  blockDim = 32 (one wave).
// Each wave computes a 32x64 tile: 2 A fragments x 4 B fragments -> 8 WMMAs
// per K-step of 32.  M % 32 == 0, N % 64 == 0, K % 32 == 0 (caller).
// ACT: 0 = none, 1 = exact GELU, 2 = ReLU.
// TDMK > 0: K == TDMK; the 64-row weight strip is staged to LDS by the
// Tensor Data Mover (one 2-D D# descriptor), B fragments then read from DS.
template <int ACT, int TDMK>
__global__ __launch_bounds__(32) void gemm_bf16_wmma(
    const bf16* __restrict__ A, const bf16* __restrict__ W,
    const float* __restrict__ bias, bf16* __restrict__ Cb,
    float* __restrict__ Cf, int M, int N, int K) {
  const int lane = threadIdx.x;
  const int half = lane >> 4;
  const int lx   = lane & 15;
  const int m0   = blockIdx.x * 32;
  const int n0   = blockIdx.y * 64;

  __shared__ unsigned short smem_raw[(TDMK > 0) ? 64 * TDMK : 1];
  const bf16* smem = (const bf16*)smem_raw;

  if (TDMK > 0) {
    // D# (ISA 8.3/8.4): 2-D tensor = this block's weight strip [64, K],
    // data_size=2B, tile = K x 64, row stride = K. One TDM op per block.
    unsigned lds = (unsigned)(size_t)(void*)smem_raw;   // LDS address (low 32 bits)
    unsigned long long ga =
        (unsigned long long)(size_t)(W + (size_t)n0 * K);
    const unsigned td0 = (unsigned)K;   // tensor_dim0 (elements)
    const unsigned td1 = 64u;           // tensor_dim1 (rows)
    v4u g0;
    g0[0] = 1u;                                              // count=1
    g0[1] = lds;                                             // lds_addr
    g0[2] = (unsigned)(ga & 0xffffffffu);                    // global_addr lo
    g0[3] = (unsigned)((ga >> 32) & 0x1ffffffu) | (2u << 30);// addr hi | type=2
    v8i g1;
    g1[0] = (int)(1u << 16);                                 // data_size = 2B
    g1[1] = (int)((td0 & 0xffffu) << 16);                    // tensor_dim0 lo
    g1[2] = (int)(((td0 >> 16) & 0xffffu) | ((td1 & 0xffffu) << 16));
    g1[3] = (int)(((td1 >> 16) & 0xffffu) | (td0 << 16));    // tile_dim0 = K
    g1[4] = (int)64u;                                        // tile_dim1 = 64
    g1[5] = (int)td0;                                        // dim0_stride lo
    g1[6] = 0;
    g1[7] = 0;
    v4i gz4 = {0, 0, 0, 0};
    v8i gz8 = {0, 0, 0, 0, 0, 0, 0, 0};
    // amdgpu-toolchain (clang-23) 6-arg form: (g0, g1, g2, g3, g4, cpol)
    __builtin_amdgcn_tensor_load_to_lds(g0, g1, gz4, gz4, gz8, 0);
    __builtin_amdgcn_s_wait_tensorcnt(0);
  }

  v8f acc[2][4];
#pragma unroll
  for (int t = 0; t < 2; ++t)
#pragma unroll
    for (int s = 0; s < 4; ++s) acc[t][s] = v8f{};

  for (int k0 = 0; k0 < K; k0 += 32) {
    v16bf a0 = load_frag16(A, K, m0 + lx, k0, half);
    v16bf a1 = load_frag16(A, K, m0 + 16 + lx, k0, half);
    if (TDMK == 0 && k0 + 128 < K) {
      // keep the A stream ahead of the WMMA pipe (global_prefetch_b8)
      __builtin_prefetch(A + (size_t)(m0 + lx) * K + k0 + 128, 0, 1);
    }
#pragma unroll
    for (int s = 0; s < 4; ++s) {
      v16bf b = (TDMK > 0)
                    ? load_frag16(smem, K, s * 16 + lx, k0, half)
                    : load_frag16(W, K, n0 + s * 16 + lx, k0, half);
      acc[0][s] = __builtin_amdgcn_wmma_f32_16x16x32_bf16(
          false, a0, false, b, (short)0, acc[0][s], false, false);
      acc[1][s] = __builtin_amdgcn_wmma_f32_16x16x32_bf16(
          false, a1, false, b, (short)0, acc[1][s], false, false);
    }
  }

  // C/D layout: lane col = lane&15 (+16*s), vgpr r -> row + r + 8*half.
#pragma unroll
  for (int t = 0; t < 2; ++t) {
#pragma unroll
    for (int s = 0; s < 4; ++s) {
      const int col = n0 + s * 16 + lx;
      const float bv = bias ? bias[col] : 0.0f;
#pragma unroll
      for (int r = 0; r < 8; ++r) {
        const int m = m0 + t * 16 + r + 8 * half;
        float v = acc[t][s][r] + bv;
        if (ACT == 1) v = gelu_f(v);
        else if (ACT == 2) v = fmaxf(v, 0.0f);
        if (Cb) Cb[(size_t)m * N + col] = f2bf(v);
        if (Cf) Cf[(size_t)m * N + col] = v;
      }
    }
  }
}

// ----------------------------- support kernels -----------------------------

// fp32 weight [rows,K] -> bf16 [rows,Kpad], zero padded.
__global__ void convert_pad(const float* __restrict__ src, bf16* __restrict__ dst,
                            int rows, int K, int Kpad) {
  long long idx = (long long)blockIdx.x * 256 + threadIdx.x;
  long long total = (long long)rows * Kpad;
  if (idx >= total) return;
  int k = (int)(idx % Kpad);
  long long r = idx / Kpad;
  float v = (k < K) ? src[r * K + k] : 0.0f;
  dst[idx] = f2bf(v);
}

// bitplanes [512,30,8,8] f32 -> patches [32768, 288] bf16 (K=270 zero padded).
__global__ void im2col_conv1(const float* __restrict__ in, bf16* __restrict__ out) {
  int idx = blockIdx.x * 256 + threadIdx.x;
  const int total = 32768 * 288;
  if (idx >= total) return;
  int k = idx % 288, row = idx / 288;
  float val = 0.0f;
  if (k < 270) {
    int c = k / 9, rem = k % 9, kh = rem / 3, kw = rem % 3;
    int b = row >> 6, pos = row & 63, h = pos >> 3, w = pos & 7;
    int ih = h + kh - 1, iw = w + kw - 1;
    if ((unsigned)ih < 8u && (unsigned)iw < 8u)
      val = in[(((size_t)b * 30 + c) * 8 + ih) * 8 + iw];
  }
  out[idx] = f2bf(val);
}

// NHWC bf16 [512*64, 256] -> patches [32768, 2304] bf16.
__global__ void im2col_conv2(const bf16* __restrict__ x, bf16* __restrict__ out) {
  long long idx = (long long)blockIdx.x * 256 + threadIdx.x;
  const long long total = 32768LL * 2304;
  if (idx >= total) return;
  int k = (int)(idx % 2304);
  int row = (int)(idx / 2304);
  int c = k / 9, rem = k % 9, kh = rem / 3, kw = rem % 3;
  int b = row >> 6, pos = row & 63, h = pos >> 3, w = pos & 7;
  int ih = h + kh - 1, iw = w + kw - 1;
  bf16 v = f2bf(0.0f);
  if ((unsigned)ih < 8u && (unsigned)iw < 8u)
    v = x[((size_t)(b * 64 + ih * 8 + iw)) * 256 + c];
  out[idx] = v;
}

// y = gelu(x * scale[c] + shift[c]); x already includes the conv bias.
__global__ void bn_gelu(const bf16* __restrict__ x, const float* __restrict__ g,
                        const float* __restrict__ bb, const float* __restrict__ mean,
                        const float* __restrict__ var, bf16* __restrict__ out,
                        long long total) {
  long long idx = (long long)blockIdx.x * 256 + threadIdx.x;
  if (idx >= total) return;
  int c = (int)(idx & 255);
  float sc = g[c] * rsqrtf(var[c] + 1e-5f);
  float sh = bb[c] - mean[c] * sc;
  out[idx] = f2bf(gelu_f(bf2f(x[idx]) * sc + sh));
}

// One wave per row of 256: out = LN(x (+ y)) * g + b.  y may be null.
__global__ __launch_bounds__(32) void ln_row(
    const bf16* __restrict__ x, const bf16* __restrict__ y,
    const float* __restrict__ g, const float* __restrict__ b,
    bf16* __restrict__ out) {
  const int row = blockIdx.x, lane = threadIdx.x;
  const bf16* xr = x + (size_t)row * 256;
  const bf16* yr = y ? y + (size_t)row * 256 : nullptr;
  float v[8], s = 0.0f;
#pragma unroll
  for (int i = 0; i < 8; ++i) {
    int c = lane * 8 + i;
    float t = bf2f(xr[c]) + (yr ? bf2f(yr[c]) : 0.0f);
    v[i] = t; s += t;
  }
  for (int o = 16; o; o >>= 1) s += __shfl_xor(s, o, 32);
  float m = s * (1.0f / 256.0f), q = 0.0f;
#pragma unroll
  for (int i = 0; i < 8; ++i) { float d = v[i] - m; q += d * d; }
  for (int o = 16; o; o >>= 1) q += __shfl_xor(q, o, 32);
  float inv = rsqrtf(q * (1.0f / 256.0f) + 1e-5f);
#pragma unroll
  for (int i = 0; i < 8; ++i) {
    int c = lane * 8 + i;
    out[(size_t)row * 256 + c] = f2bf((v[i] - m) * inv * g[c] + b[c]);
  }
}

// One wave per (b, head, q): softmax(q.K^T / 8) . V, dh = 64, S in {64,3,2}.
__global__ __launch_bounds__(32) void attention_kernel(
    const bf16* __restrict__ qkv, bf16* __restrict__ out, int S) {
  const int lane = threadIdx.x;
  int idx = blockIdx.x;
  int q = idx % S; int t = idx / S; int h = t & 3; int b = t >> 2;
  const size_t rowQ = (size_t)(b * S + q) * 768 + h * 64;
  __shared__ float qv[64];
  __shared__ float sc[64];
  qv[lane]      = bf2f(qkv[rowQ + lane]);
  qv[lane + 32] = bf2f(qkv[rowQ + lane + 32]);
  __syncthreads();
  float s0 = -1e30f, s1 = -1e30f;
#pragma unroll
  for (int j = 0; j < 2; ++j) {
    int k = lane + j * 32;
    if (k < S) {
      const bf16* kr = qkv + (size_t)(b * S + k) * 768 + 256 + h * 64;
      float d = 0.0f;
      for (int e = 0; e < 64; ++e) d += qv[e] * bf2f(kr[e]);
      if (j == 0) s0 = d * 0.125f; else s1 = d * 0.125f;
    }
  }
  float mx = fmaxf(s0, s1);
  for (int o = 16; o; o >>= 1) mx = fmaxf(mx, __shfl_xor(mx, o, 32));
  float p0 = (s0 > -1e29f) ? __expf(s0 - mx) : 0.0f;
  float p1 = (s1 > -1e29f) ? __expf(s1 - mx) : 0.0f;
  float sum = p0 + p1;
  for (int o = 16; o; o >>= 1) sum += __shfl_xor(sum, o, 32);
  float inv = 1.0f / sum;
  sc[lane] = p0 * inv; sc[lane + 32] = p1 * inv;
  __syncthreads();
#pragma unroll
  for (int j = 0; j < 2; ++j) {
    int d = lane + j * 32;
    float acc = 0.0f;
    for (int k = 0; k < S; ++k) {
      const bf16* vr = qkv + (size_t)(b * S + k) * 768 + 512 + h * 64;
      acc += sc[k] * bf2f(vr[d]);
    }
    out[(size_t)(b * S + q) * 256 + h * 64 + d] = f2bf(acc);
  }
}

__global__ void stack3(const bf16* __restrict__ a, const bf16* __restrict__ b,
                       const bf16* __restrict__ c, bf16* __restrict__ out) {
  int idx = blockIdx.x * 256 + threadIdx.x;
  if (idx >= BATCHC * 3 * 256) return;
  int col = idx & 255; int s = (idx >> 8) % 3; int bb = idx / (3 * 256);
  const bf16* src = (s == 0) ? a : ((s == 1) ? b : c);
  out[idx] = src[bb * 256 + col];
}

__global__ void stack2(const bf16* __restrict__ a, const bf16* __restrict__ b,
                       bf16* __restrict__ out) {
  int idx = blockIdx.x * 256 + threadIdx.x;
  if (idx >= BATCHC * 2 * 256) return;
  int col = idx & 255; int s = (idx >> 8) % 2; int bb = idx / (2 * 256);
  const bf16* src = (s == 0) ? a : b;
  out[idx] = src[bb * 256 + col];
}

__global__ void mean_tokens(const bf16* __restrict__ in, bf16* __restrict__ out, int S) {
  int idx = blockIdx.x * 256 + threadIdx.x;
  if (idx >= BATCHC * 256) return;
  int b = idx >> 8, c = idx & 255;
  float s = 0.0f;
  for (int t = 0; t < S; ++t) s += bf2f(in[((size_t)(b * S + t)) * 256 + c]);
  out[idx] = f2bf(s / (float)S);
}

__global__ void fill_zero_bf16(bf16* __restrict__ p, int n) {
  int idx = blockIdx.x * 256 + threadIdx.x;
  if (idx < n) p[idx] = f2bf(0.0f);
}

// ----------------------------- host orchestration --------------------------

struct EncW {
  const bf16 *in_w, *out_w, *w1, *w2;
  const float *in_b, *out_b, *b1, *b2, *ln1g, *ln1b, *ln2g, *ln2b;
};

extern "C" void kernel_launch(void* const* d_in, const int* in_sizes, int n_in,
                              void* d_out, int out_size, void* d_ws, size_t ws_size,
                              hipStream_t stream) {
  (void)in_sizes; (void)out_size; (void)ws_size;
  const float* F[64];
  for (int i = 0; i < n_in && i < 64; ++i) F[i] = (const float*)d_in[i];
  // Input order (depth-first dict order of setup_inputs):
  // 0 bitplanes | 1 cw1 2 cb1 3-6 bn1(g,b,mean,var) 7 cw2 8 cb2 9-12 bn2
  // 13-24 attn(in_w,in_b,out_w,out_b,w1,b1,w2,b2,ln1g,ln1b,ln2g,ln2b)
  // 25-30 enh(w1,b1,w2,b2,ln_g,ln_b) | 31-42 L | 43-54 H | 55-58 head

  char* base = (char*)d_ws;
  size_t off = 0;
  auto alloc = [&](size_t nelem) -> bf16* {
    bf16* p = (bf16*)(base + off);
    off += (nelem * sizeof(bf16) + 255) & ~(size_t)255;
    return p;
  };

  // bf16 weights (converted every call; deterministic, no caching)
  bf16* wc1  = alloc(256ULL * 288);
  bf16* wc2  = alloc(256ULL * 2304);
  bf16 *wA[4], *wL[4], *wH[4];
  for (int e = 0; e < 3; ++e) {
    bf16** w = (e == 0) ? wA : (e == 1) ? wL : wH;
    w[0] = alloc(768ULL * 256);  // in_w
    w[1] = alloc(256ULL * 256);  // out_w
    w[2] = alloc(512ULL * 256);  // w1
    w[3] = alloc(256ULL * 512);  // w2
  }
  bf16* we1 = alloc(1024ULL * 16384);
  bf16* we2 = alloc(256ULL * 1024);
  bf16* wh1 = alloc(256ULL * 256);
  bf16* wh2 = alloc(128ULL * 256);

  // persistent activations
  bf16* x    = alloc(32768ULL * 256);  // token stream (conv out / residual)
  bf16* xpre = alloc(32768ULL * 256);  // conv2 pre-BN
  bf16* x2   = alloc(32768ULL * 256);  // encoder out == [512,16384] flat
  bf16* be   = alloc(512ULL * 256);
  bf16* zL   = alloc(512ULL * 256);
  bf16* zH   = alloc(512ULL * 256);
  // HRM micro-encoder buffers (sized for M = 1536)
  bf16* lin  = alloc(1536ULL * 256);
  bf16* lqkv = alloc(1536ULL * 768);
  bf16* lat  = alloc(1536ULL * 256);
  bf16* lx1  = alloc(1536ULL * 256);
  bf16* lff  = alloc(1536ULL * 512);
  bf16* lt1  = alloc(1536ULL * 256);
  bf16* lx2  = alloc(1536ULL * 256);
  bf16* hh   = alloc(512ULL * 256);
  bf16* h1   = alloc(512ULL * 1024);
  bf16* h2   = alloc(512ULL * 256);
  // big time-shared scratch (151 MB): im2col2 OR big-encoder temporaries
  bf16* scratch = alloc(32768ULL * 2304);
  bf16* A1    = scratch;
  bf16* t0    = scratch + 32768ULL * 288;
  bf16* A2    = scratch;
  bf16* qkv   = scratch;
  bf16* attnT = scratch + 32768ULL * 768;
  bf16* x1    = attnT + 32768ULL * 256;
  bf16* ffT   = x1 + 32768ULL * 256;
  bf16* t1    = ffT + 32768ULL * 512;

  auto ew = [](long long n) { return dim3((unsigned)((n + 255) / 256)); };
  auto cvt = [&](const float* s, bf16* d, int rows, int K, int Kpad) {
    convert_pad<<<ew((long long)rows * Kpad), 256, 0, stream>>>(s, d, rows, K, Kpad);
  };
  // GEMM dispatch: TDM-staged LDS weights for K in {256, 288, 512},
  // streaming global weights otherwise.
  auto gemm = [&](const bf16* A, const bf16* W, const float* bias, bf16* Cb,
                  float* Cf, int M, int N, int K, int act) {
    dim3 g(M / 32, N / 64);
#define GL(KERN) KERN<<<g, 32, 0, stream>>>(A, W, bias, Cb, Cf, M, N, K)
    if (K == 256) {
      if (act == 1)      GL((gemm_bf16_wmma<1, 256>));
      else if (act == 2) GL((gemm_bf16_wmma<2, 256>));
      else               GL((gemm_bf16_wmma<0, 256>));
    } else if (K == 512) {
      GL((gemm_bf16_wmma<0, 512>));          // only act==0 uses K==512
    } else if (K == 288) {
      GL((gemm_bf16_wmma<0, 288>));          // conv1
    } else {
      if (act == 1)      GL((gemm_bf16_wmma<1, 0>));
      else if (act == 2) GL((gemm_bf16_wmma<2, 0>));
      else               GL((gemm_bf16_wmma<0, 0>));
    }
#undef GL
  };
  auto run_enc = [&](const bf16* xin, bf16* q, bf16* at, bf16* xx1, bf16* ff,
                     bf16* tt, bf16* xo, int Bb, int S, const EncW& w, int act) {
    const int M = Bb * S;
    gemm(xin, w.in_w, w.in_b, q, nullptr, M, 768, 256, 0);
    attention_kernel<<<dim3(Bb * NHEADC * S), 32, 0, stream>>>(q, at, S);
    gemm(at, w.out_w, w.out_b, tt, nullptr, M, 256, 256, 0);
    ln_row<<<dim3(M), 32, 0, stream>>>(xin, tt, w.ln1g, w.ln1b, xx1);
    gemm(xx1, w.w1, w.b1, ff, nullptr, M, DFFC, 256, act);
    gemm(ff, w.w2, w.b2, tt, nullptr, M, 256, DFFC, 0);
    ln_row<<<dim3(M), 32, 0, stream>>>(xx1, tt, w.ln2g, w.ln2b, xo);
  };

  // ---- weight conversion (fp32 -> bf16, zero pad conv1 K 270 -> 288)
  cvt(F[1], wc1, 256, 270, 288);
  cvt(F[7], wc2, 256, 2304, 2304);
  const int encBase[3] = {13, 31, 43};
  for (int e = 0; e < 3; ++e) {
    bf16** w = (e == 0) ? wA : (e == 1) ? wL : wH;
    int b0 = encBase[e];
    cvt(F[b0 + 0], w[0], 768, 256, 256);
    cvt(F[b0 + 2], w[1], 256, 256, 256);
    cvt(F[b0 + 4], w[2], 512, 256, 256);
    cvt(F[b0 + 6], w[3], 256, 512, 512);
  }
  cvt(F[25], we1, 1024, 16384, 16384);
  cvt(F[27], we2, 256, 1024, 1024);
  cvt(F[55], wh1, 256, 256, 256);
  cvt(F[57], wh2, 128, 256, 256);

  EncW encA = {wA[0], wA[1], wA[2], wA[3], F[14], F[16], F[18], F[20],
               F[21], F[22], F[23], F[24]};
  EncW encL = {wL[0], wL[1], wL[2], wL[3], F[32], F[34], F[36], F[38],
               F[39], F[40], F[41], F[42]};
  EncW encH = {wH[0], wH[1], wH[2], wH[3], F[44], F[46], F[48], F[50],
               F[51], F[52], F[53], F[54]};

  // ---- conv front-end (im2col + WMMA GEMM, NHWC)
  im2col_conv1<<<ew(32768LL * 288), 256, 0, stream>>>(F[0], A1);
  gemm(A1, wc1, F[2], t0, nullptr, 32768, 256, 288, 0);
  bn_gelu<<<ew(32768LL * 256), 256, 0, stream>>>(t0, F[3], F[4], F[5], F[6], x,
                                                 32768LL * 256);
  im2col_conv2<<<ew(32768LL * 2304), 256, 0, stream>>>(x, A2);
  gemm(A2, wc2, F[8], xpre, nullptr, 32768, 256, 2304, 0);
  bn_gelu<<<ew(32768LL * 256), 256, 0, stream>>>(xpre, F[9], F[10], F[11], F[12],
                                                 x, 32768LL * 256);

  // ---- big transformer encoder over [512, 64, 256] (GELU FF)
  run_enc(x, qkv, attnT, x1, ffT, t1, x2, BATCHC, SEQC, encA, 1);

  // ---- enhancer MLP on flattened [512, 16384]
  gemm(x2, we1, F[26], h1, nullptr, 512, 1024, 16384, 1);
  gemm(h1, we2, F[28], h2, nullptr, 512, 256, 1024, 1);
  ln_row<<<dim3(512), 32, 0, stream>>>(h2, nullptr, F[29], F[30], be);

  // ---- HRM loop: 63 scan steps + final L and H step
  fill_zero_bf16<<<ew(512 * 256), 256, 0, stream>>>(zL, 512 * 256);
  fill_zero_bf16<<<ew(512 * 256), 256, 0, stream>>>(zH, 512 * 256);
  auto l_step = [&]() {
    stack3<<<ew(512LL * 3 * 256), 256, 0, stream>>>(zL, zH, be, lin);
    run_enc(lin, lqkv, lat, lx1, lff, lt1, lx2, BATCHC, 3, encL, 2);
    mean_tokens<<<ew(512 * 256), 256, 0, stream>>>(lx2, zL, 3);
  };
  auto h_step = [&]() {
    stack2<<<ew(512LL * 2 * 256), 256, 0, stream>>>(zH, zL, lin);
    run_enc(lin, lqkv, lat, lx1, lff, lt1, lx2, BATCHC, 2, encH, 2);
    mean_tokens<<<ew(512 * 256), 256, 0, stream>>>(lx2, zH, 2);
  };
  for (int i = 0; i < 63; ++i) {  // N*T - 1 scan steps
    l_step();
    if ((i + 1) % 8 == 0) h_step();
  }
  l_step();
  h_step();

  // ---- head: [512,256] -> GELU -> [512,128] f32 out
  gemm(zH, wh1, F[56], hh, nullptr, 512, 256, 256, 1);
  gemm(hh, wh2, F[58], nullptr, (float*)d_out, 512, 128, 256, 0);
}
